// TopKRouter_18356690223396
// MI455X (gfx1250) — matrix operand order repaired and checked by reading, not compiled, not run on hardware
//
#include <hip/hip_runtime.h>

// ---------------------------------------------------------------------------
// TopK router: logits = x(16384x2048) @ W(2048x64) + b; softmax over 64; top-2.
// bf16 WMMA (v_wmma_f32_16x16x32_bf16) with fp32 accumulation.
// x stream (134 MB) is the roofline floor (~5.8us @ 23.3 TB/s); bf16 WMMA keeps
// compute far below that; W is repacked once to bf16 fragments living in L2.
// ---------------------------------------------------------------------------

typedef __attribute__((ext_vector_type(16))) __bf16 v16bf;
typedef __attribute__((ext_vector_type(8)))  float  v8f;
typedef __attribute__((ext_vector_type(8)))  unsigned int v8u;

#define D_DIM   2048
#define E_DIM   64
#define KCHUNK  32
#define NKC     (D_DIM / KCHUNK)          // 64 K-chunks
#define PFDIST  8                          // prefetch distance in K-chunks
#define WAVES_PER_BLOCK 8
#define ROWS_PER_WAVE   16
#define ROWS_PER_BLOCK  (WAVES_PER_BLOCK * ROWS_PER_WAVE)  // 128

// Pack two fp32 -> one dword of 2 bf16 by truncation (single v_perm_b32).
__device__ __forceinline__ unsigned pack_bf16_trunc(float lo, float hi) {
  // D[15:0] = lo[31:16], D[31:16] = hi[31:16]
  return __builtin_amdgcn_perm(__builtin_bit_cast(unsigned, hi),
                               __builtin_bit_cast(unsigned, lo),
                               0x07060302u);
}

__device__ __forceinline__ unsigned short f32_to_bf16_rne(float f) {
  unsigned u = __builtin_bit_cast(unsigned, f);
  unsigned r = u + 0x7FFFu + ((u >> 16) & 1u);
  return (unsigned short)(r >> 16);
}

// ---------------------------------------------------------------------------
// Prep: repack W (D x E fp32, row-major) into bf16 B-fragments in d_ws.
// Fragment element (kc, t, lane, j) -> W[(kc*32 + (j>=8?16:0) + (lane>=16?8:0) + (j&7))*64 + t*16 + (lane&15)]
// Layout: frag[((kc*4 + t)*32 + lane)*16 + j]  => per-wave loads are contiguous.
// ---------------------------------------------------------------------------
__global__ __launch_bounds__(256) void prep_W_kernel(const float* __restrict__ W,
                                                     unsigned short* __restrict__ frag) {
  int e = blockIdx.x * 256 + threadIdx.x;      // 0 .. 131071
  int j    = e & 15;
  int lane = (e >> 4) & 31;
  int t    = (e >> 9) & 3;
  int kc   = e >> 11;
  int col  = t * 16 + (lane & 15);
  int k    = kc * KCHUNK + ((j >> 3) << 4) + ((lane >> 4) << 3) + (j & 7);
  frag[e] = f32_to_bf16_rne(W[k * E_DIM + col]);
}

// One K-chunk of the GEMM body. PREFETCH toggles the x-stream prefetch so the
// tail loop carries no per-iteration branch.
template <bool PREFETCH>
__device__ __forceinline__ void kstep(int kc, const float* __restrict__ xrow,
                                      const unsigned short* __restrict__ wbase,
                                      v8f& acc0, v8f& acc1, v8f& acc2, v8f& acc3) {
  const float* ap = xrow + kc * KCHUNK;
  float4 a0 = *(const float4*)(ap + 0);
  float4 a1 = *(const float4*)(ap + 4);
  float4 a2 = *(const float4*)(ap + 16);
  float4 a3 = *(const float4*)(ap + 20);

  if (PREFETCH) __builtin_prefetch(ap + PFDIST * KCHUNK, 0, 1);

  v8u apk;
  apk[0] = pack_bf16_trunc(a0.x, a0.y);
  apk[1] = pack_bf16_trunc(a0.z, a0.w);
  apk[2] = pack_bf16_trunc(a1.x, a1.y);
  apk[3] = pack_bf16_trunc(a1.z, a1.w);
  apk[4] = pack_bf16_trunc(a2.x, a2.y);
  apk[5] = pack_bf16_trunc(a2.z, a2.w);
  apk[6] = pack_bf16_trunc(a3.x, a3.y);
  apk[7] = pack_bf16_trunc(a3.z, a3.w);
  v16bf A = __builtin_bit_cast(v16bf, apk);

  const unsigned short* wb = wbase + (size_t)kc * (4 * 512);
  v16bf B0 = __builtin_bit_cast(v16bf, *(const v8u*)(wb + 0));
  v16bf B1 = __builtin_bit_cast(v16bf, *(const v8u*)(wb + 512));
  v16bf B2 = __builtin_bit_cast(v16bf, *(const v8u*)(wb + 1024));
  v16bf B3 = __builtin_bit_cast(v16bf, *(const v8u*)(wb + 1536));

  acc0 = __builtin_amdgcn_wmma_f32_16x16x32_bf16(false, A, false, B0, (short)0, acc0, false, false);
  acc1 = __builtin_amdgcn_wmma_f32_16x16x32_bf16(false, A, false, B1, (short)0, acc1, false, false);
  acc2 = __builtin_amdgcn_wmma_f32_16x16x32_bf16(false, A, false, B2, (short)0, acc2, false, false);
  acc3 = __builtin_amdgcn_wmma_f32_16x16x32_bf16(false, A, false, B3, (short)0, acc3, false, false);
}

// ---------------------------------------------------------------------------
// Main kernel: GEMM (bf16 WMMA) + bias + softmax + top-2, fully fused.
// ---------------------------------------------------------------------------
__global__ __launch_bounds__(256) void router_topk_kernel(
    const float* __restrict__ x,
    const float* __restrict__ bias,
    const unsigned short* __restrict__ wfrag,
    float* __restrict__ out, int M) {

  const int lane = threadIdx.x & 31;
  const int wave = threadIdx.x >> 5;
  const int half = lane >> 4;       // lane half selects K sub-range
  const int lid  = lane & 15;       // row (for A) / column-in-tile (for B/C)
  const int rowBase = blockIdx.x * ROWS_PER_BLOCK + wave * ROWS_PER_WAVE;

  // A: lane (half,lid) covers row rowBase+lid; first 8 K at +half*8, next 8 at +16.
  const float* xrow = x + (size_t)(rowBase + lid) * D_DIM + half * 8;
  const unsigned short* wbase = wfrag + (size_t)lane * 16;

  v8f acc0 = {}, acc1 = {}, acc2 = {}, acc3 = {};

  // main body: branch-free inner loop with x-stream prefetch
#pragma unroll 2
  for (int kc = 0; kc < NKC - PFDIST; ++kc)
    kstep<true>(kc, xrow, wbase, acc0, acc1, acc2, acc3);
  // tail: last PFDIST chunks, no prefetch (stays in-bounds), fully unrolled
#pragma unroll
  for (int kc = NKC - PFDIST; kc < NKC; ++kc)
    kstep<false>(kc, xrow, wbase, acc0, acc1, acc2, acc3);

  // ---- bias add (per column; same for all rows) ----
  float b0 = bias[lid + 0];
  float b1 = bias[lid + 16];
  float b2 = bias[lid + 32];
  float b3 = bias[lid + 48];
#pragma unroll
  for (int r = 0; r < 8; ++r) {
    acc0[r] += b0; acc1[r] += b1; acc2[r] += b2; acc3[r] += b3;
  }

  // ---- per-row softmax + top-2 ----
  // C layout: reg r, lanes 0-15 = row r (cols lid+16t); lanes 16-31 = row r+8.
  const float LOG2E = 1.4426950408889634f;
  const size_t idxRegion = (size_t)M * 2;   // first M*2 floats = indices, next M*2 = probs

#pragma unroll
  for (int r = 0; r < 8; ++r) {
    float v0 = acc0[r], v1 = acc1[r], v2 = acc2[r], v3 = acc3[r];
    int   c0 = lid, c1 = lid + 16, c2 = lid + 32, c3 = lid + 48;

    // local top-2 of 4 candidates (ties -> lower index; indices ascending)
    float t1v, t2v; int t1i, t2i;
    if (v1 > v0) { t1v = v1; t1i = c1; t2v = v0; t2i = c0; }
    else         { t1v = v0; t1i = c0; t2v = v1; t2i = c1; }
    if (v2 > t1v) { t2v = t1v; t2i = t1i; t1v = v2; t1i = c2; }
    else if (v2 > t2v) { t2v = v2; t2i = c2; }
    if (v3 > t1v) { t2v = t1v; t2i = t1i; t1v = v3; t1i = c3; }
    else if (v3 > t2v) { t2v = v3; t2i = c3; }

    // butterfly merge across the 16-lane half (masks 1,2,4,8 stay within half)
#pragma unroll
    for (int m = 1; m <= 8; m <<= 1) {
      float o1v = __shfl_xor(t1v, m, 32);
      int   o1i = __shfl_xor(t1i, m, 32);
      float o2v = __shfl_xor(t2v, m, 32);
      int   o2i = __shfl_xor(t2i, m, 32);
      bool oFirst = (o1v > t1v) || (o1v == t1v && o1i < t1i);
      float n1v = oFirst ? o1v : t1v;  int n1i = oFirst ? o1i : t1i;
      float cav = oFirst ? t1v : o1v;  int cai = oFirst ? t1i : o1i; // loser of firsts
      float cbv = oFirst ? o2v : t2v;  int cbi = oFirst ? o2i : t2i; // winner's second
      bool aSec = (cav > cbv) || (cav == cbv && cai < cbi);
      t1v = n1v; t1i = n1i;
      t2v = aSec ? cav : cbv;
      t2i = aSec ? cai : cbi;
    }

    // softmax denominator with max = t1v
    float mx = t1v;
    float s = __builtin_amdgcn_exp2f((v0 - mx) * LOG2E)
            + __builtin_amdgcn_exp2f((v1 - mx) * LOG2E)
            + __builtin_amdgcn_exp2f((v2 - mx) * LOG2E)
            + __builtin_amdgcn_exp2f((v3 - mx) * LOG2E);
#pragma unroll
    for (int m = 1; m <= 8; m <<= 1) s += __shfl_xor(s, m, 32);

    if (lid == 0) {
      int row = rowBase + r + half * 8;
      float inv = 1.0f / s;
      float p1 = inv;  // exp(t1v - mx) == 1
      float p2 = __builtin_amdgcn_exp2f((t2v - mx) * LOG2E) * inv;
      out[(size_t)row * 2 + 0] = (float)t1i;
      out[(size_t)row * 2 + 1] = (float)t2i;
      out[idxRegion + (size_t)row * 2 + 0] = p1;
      out[idxRegion + (size_t)row * 2 + 1] = p2;
    }
  }
}

// ---------------------------------------------------------------------------
extern "C" void kernel_launch(void* const* d_in, const int* in_sizes, int n_in,
                              void* d_out, int out_size, void* d_ws, size_t ws_size,
                              hipStream_t stream) {
  const float* x = (const float*)d_in[0];
  const float* W = (const float*)d_in[1];
  const float* b = (const float*)d_in[2];
  float* out = (float*)d_out;

  int M = in_sizes[0] / D_DIM;              // 16384 rows
  unsigned short* wfrag = (unsigned short*)d_ws;  // needs 256 KB

  // repack W -> bf16 fragments (131072 elements)
  prep_W_kernel<<<(D_DIM * E_DIM) / 256, 256, 0, stream>>>(W, wfrag);

  // fused GEMM + softmax + top-2
  int blocks = M / ROWS_PER_BLOCK;          // 128
  router_topk_kernel<<<blocks, 256, 0, stream>>>(x, b, wfrag, out, M);
}
